// PNA_9826885173934
// MI455X (gfx1250) — matrix operand order repaired and checked by reading, not compiled, not run on hardware
//
#include <hip/hip_runtime.h>
#include <hip/hip_bf16.h>
#include <math.h>

// ---------------- problem constants ----------------
#define N_NODES  50000
#define N_EDGES  500000
#define HID      64
#define NLAYERS  4
#define DIN_U    1216      // (6*3+1)*64
#define KT_M     4         // 128/32 k-steps for message GEMM
#define KT_U     38        // 1216/32 k-steps for update GEMM
#define KT_X     2         // 64/32  k-steps for mix GEMM
#define DELTA_S  2.5f
#define EPS_BN   1e-5f

// packed fragment bytes per layer for each weight
#define WM_FRAG_BYTES  ((size_t)KT_M * 4 * 32 * 16 * 2)   // 16 KB
#define WU_FRAG_BYTES  ((size_t)KT_U * 4 * 32 * 16 * 2)   // 152 KB (fits 320KB WGP LDS)
#define WX_FRAG_BYTES  ((size_t)KT_X * 4 * 32 * 16 * 2)   // 8 KB

typedef __attribute__((ext_vector_type(16))) __bf16 v16bf;
typedef __attribute__((ext_vector_type(8)))  float  v8f;

// monotone float <-> uint key so max/min become native u32 atomics
__device__ __forceinline__ unsigned fkey(float f){
  unsigned u = __float_as_uint(f);
  return (u & 0x80000000u) ? ~u : (u | 0x80000000u);
}
__device__ __forceinline__ float unfkey(unsigned k){
  return __uint_as_float((k & 0x80000000u) ? (k & 0x7FFFFFFFu) : ~k);
}

// A-fragment element e (0..15) -> local K index, contiguous: k = half*16 + e (matches B packing)
__device__ __forceinline__ int kOfA(int e, int half){
  return half * 16 + e;
}

// cooperative global->LDS copy of packed weight fragments (16B vectors)
__device__ __forceinline__ void fill_lds(void* smem, const void* gsrc, int bytes){
  uint4* d = (uint4*)smem;
  const uint4* s = (const uint4*)gsrc;
  int nvec = bytes >> 4;
  for (int i = threadIdx.x; i < nvec; i += blockDim.x) d[i] = s[i];
  __syncthreads();
}

// ---------------- weight repack: fp32 row-major -> bf16 WMMA B-fragments ----
// B 32x16 bf16 layout: lane: n = lane&15, half = lane>>4; elem e -> k = half*16 + e
// Packed so each lane's 16 bf16 (32B) are contiguous -> coalesced / LDS friendly.
__global__ void k_pack(const float* __restrict__ W, __bf16* __restrict__ P,
                       int Kdim, int nLayers){
  int g = blockIdx.x * blockDim.x + threadIdx.x;
  int KT = Kdim >> 5;
  int total = nLayers * KT * 4 * 32;
  if (g >= total) return;
  int lane = g & 31;
  int r = g >> 5;
  int nt = r & 3;  r >>= 2;
  int kt = r % KT;
  int layer = r / KT;
  int half = lane >> 4;
  int n = (lane & 15) + nt * 16;
  const float* Wl = W + (size_t)layer * Kdim * HID;
  __bf16* dst = P + (size_t)g * 16;
  for (int e = 0; e < 16; ++e){
    int k = kt * 32 + half * 16 + e;
    dst[e] = (__bf16)Wl[(size_t)k * HID + n];
  }
}

// ---------------- per-layer accumulator clear ----------------
__global__ void k_clear(float* __restrict__ sum, float* __restrict__ sumsq,
                        unsigned* __restrict__ mxk, unsigned* __restrict__ mnk,
                        float* __restrict__ deg, float* __restrict__ st1,
                        float* __restrict__ st2){
  int i = blockIdx.x * blockDim.x + threadIdx.x;
  if (i < N_NODES * HID){
    sum[i]   = 0.f;
    sumsq[i] = 0.f;
    mxk[i]   = 0u;            // key(-inf) lower bound
    mnk[i]   = 0xFFFFFFFFu;   // key(+inf) upper bound
  }
  if (i < N_NODES) deg[i] = 0.f;
  if (i < 128){ st1[i] = 0.f; st2[i] = 0.f; }
}

// ---------------- edge messages (WMMA) + scatter aggregation ----------------
// One wave = 16 edges. msg = [h_src || h_dst] @ WM + bM, then atomics into
// per-node sum / sumsq / maxkey / minkey, plus degree count.
// WM fragments staged once per block in LDS (16 KB).
__global__ void k_edge(const float* __restrict__ h,
                       const int* __restrict__ esrc, const int* __restrict__ edst,
                       const __bf16* __restrict__ WMp, const float* __restrict__ bM,
                       float* __restrict__ sum, float* __restrict__ sumsq,
                       unsigned* __restrict__ mxk, unsigned* __restrict__ mnk,
                       float* __restrict__ deg){
  extern __shared__ char smem_raw[];
  __bf16* sW = (__bf16*)smem_raw;
  fill_lds(sW, WMp, (int)WM_FRAG_BYTES);      // single barrier, before any guard

  int lane = threadIdx.x & 31;
  int wid  = threadIdx.x >> 5;
  int tile = blockIdx.x * 8 + wid;            // 16-edge tile
  long e0 = (long)tile * 16;
  if (e0 >= N_EDGES) return;                  // safe: no barriers after this
  int half = lane >> 4;
  int mrow = lane & 15;
  int se = esrc[e0 + mrow];
  int de = edst[e0 + mrow];
  if (lane < 16) atomicAdd(&deg[de], 1.0f);

  const float* hs = h + (size_t)se * HID;
  const float* hd = h + (size_t)de * HID;

  v8f zero = {};
  v8f acc[4]; for (int t = 0; t < 4; ++t) acc[t] = zero;

  for (int kt = 0; kt < KT_M; ++kt){
    v16bf a;
    for (int e = 0; e < 16; ++e){
      int k = kt * 32 + kOfA(e, half);
      float v = (k < HID) ? hs[k] : hd[k - HID];
      a[e] = (__bf16)v;
    }
    for (int nt = 0; nt < 4; ++nt){
      const v16bf* bp =
        (const v16bf*)(sW + (((size_t)kt * 4 + nt) * 32 + lane) * 16);
      acc[nt] = __builtin_amdgcn_wmma_f32_16x16x32_bf16(
          false, a, false, *bp, (short)0, acc[nt], false, false);
    }
  }

  int dsts[8];
  for (int r = 0; r < 8; ++r) dsts[r] = edst[e0 + r + 8 * half];

  for (int nt = 0; nt < 4; ++nt){
    int n = (lane & 15) + nt * 16;
    float bb = bM[n];
    for (int r = 0; r < 8; ++r){
      float v = acc[nt][r] + bb;
      size_t off = (size_t)dsts[r] * HID + n;
      atomicAdd(&sum[off], v);
      atomicAdd(&sumsq[off], v * v);
      atomicMax(&mxk[off], fkey(v));
      atomicMin(&mnk[off], fkey(v));
    }
  }
}

// ---------------- node update GEMM (WMMA, K=1216) ----------------
// Entire per-layer WU fragment block (152 KB bf16) held in WGP LDS (CDNA5:
// 320 KB per workgroup), shared by all 8 waves of the block.
// A row built on the fly: [h || agg || agg*amp || agg*att].
__global__ void k_update(const float* __restrict__ h,
                         const float* __restrict__ sum, const float* __restrict__ sumsq,
                         const unsigned* __restrict__ mxk, const unsigned* __restrict__ mnk,
                         const float* __restrict__ deg,
                         const __bf16* __restrict__ WUp, const float* __restrict__ bU,
                         float* __restrict__ hu){
  extern __shared__ char smem_raw[];
  __bf16* sW = (__bf16*)smem_raw;
  fill_lds(sW, WUp, (int)WU_FRAG_BYTES);

  int lane = threadIdx.x & 31;
  int wid  = threadIdx.x >> 5;
  int tile = blockIdx.x * 8 + wid;            // 16-node tile
  if (tile * 16 >= N_NODES) return;
  int half = lane >> 4;
  int nd = tile * 16 + (lane & 15);

  float dg    = deg[nd];
  float inv   = 1.f / fmaxf(dg, 1.f);
  bool  has   = dg > 0.f;
  float logD  = logf(dg + 1.f);
  float scl1  = logD * (1.f / DELTA_S);
  float scl2  = (logD > 0.f) ? (DELTA_S / logD) : 0.f;

  const float*    hrow = h     + (size_t)nd * HID;
  const float*    srow = sum   + (size_t)nd * HID;
  const float*    qrow = sumsq + (size_t)nd * HID;
  const unsigned* xrow = mxk   + (size_t)nd * HID;
  const unsigned* nrow = mnk   + (size_t)nd * HID;

  v8f zero = {};
  v8f acc[4]; for (int t = 0; t < 4; ++t) acc[t] = zero;

  for (int kt = 0; kt < KT_U; ++kt){
    v16bf a;
    for (int e = 0; e < 16; ++e){
      int k = kt * 32 + kOfA(e, half);
      float v;
      if (k < HID){
        v = hrow[k];
      } else {
        int t2 = k - HID;
        int s  = t2 / (6 * HID);          // scaler: 0 ident, 1 amp, 2 att
        int j  = t2 - s * (6 * HID);
        int ai = j >> 6;                  // 0 mean,1 max,2 min,3 std,4 var,5 sum
        int f  = j & 63;
        float base;
        if (ai == 0)      base = srow[f] * inv;
        else if (ai == 1) base = has ? unfkey(xrow[f]) : 0.f;
        else if (ai == 2) base = has ? unfkey(nrow[f]) : 0.f;
        else if (ai == 5) base = srow[f];
        else {
          float mean = srow[f] * inv;
          float var  = fmaxf(qrow[f] * inv - mean * mean, 0.f);
          base = (ai == 4) ? var : sqrtf(var + 1e-30f);
        }
        float sc = (s == 0) ? 1.f : (s == 1 ? scl1 : scl2);
        v = base * sc;
      }
      a[e] = (__bf16)v;
    }
    for (int nt = 0; nt < 4; ++nt){
      const v16bf* bp =
        (const v16bf*)(sW + (((size_t)kt * 4 + nt) * 32 + lane) * 16);
      acc[nt] = __builtin_amdgcn_wmma_f32_16x16x32_bf16(
          false, a, false, *bp, (short)0, acc[nt], false, false);
    }
  }

  const float invSqrtN = rsqrtf((float)N_NODES);  // dgl graph-size norm
  for (int nt = 0; nt < 4; ++nt){
    int n = (lane & 15) + nt * 16;
    float bb = bU[n];
    for (int r = 0; r < 8; ++r){
      int node = tile * 16 + r + 8 * half;
      hu[(size_t)node * HID + n] = (acc[nt][r] + bb) * invSqrtN;
    }
  }
}

// ---------------- per-column sum / sumsq over all nodes ----------------
__global__ void k_colstats(const float* __restrict__ x, float* __restrict__ st){
  int tid  = threadIdx.x;
  int col  = tid & 63;
  int row0 = blockIdx.x * 4 + (tid >> 6);
  float s = 0.f, q = 0.f;
  for (int r = row0; r < N_NODES; r += 256){
    float v = x[(size_t)r * HID + col];
    s += v; q += v * v;
  }
  atomicAdd(&st[col], s);
  atomicAdd(&st[64 + col], q);
}

// ---------------- BN(tower) + mix GEMM (WMMA) + leaky + residual + stats ----
__global__ void k_mix(const float* __restrict__ hu, const float* __restrict__ st1,
                      const float* __restrict__ g_t, const float* __restrict__ b_t,
                      const __bf16* __restrict__ WXp, const float* __restrict__ bmix,
                      const float* __restrict__ h, float* __restrict__ out,
                      float* __restrict__ st2){
  extern __shared__ char smem_raw[];
  __bf16* sW = (__bf16*)smem_raw;
  fill_lds(sW, WXp, (int)WX_FRAG_BYTES);

  int lane = threadIdx.x & 31;
  int wid  = threadIdx.x >> 5;
  int tile = blockIdx.x * 8 + wid;
  if (tile * 16 >= N_NODES) return;
  int half = lane >> 4;
  int nd = tile * 16 + (lane & 15);
  const float invN = 1.f / (float)N_NODES;

  v8f zero = {};
  v8f acc[4]; for (int t = 0; t < 4; ++t) acc[t] = zero;

  for (int kt = 0; kt < KT_X; ++kt){
    v16bf a;
    for (int e = 0; e < 16; ++e){
      int k = kt * 32 + kOfA(e, half);
      float x  = hu[(size_t)nd * HID + k];
      float mu = st1[k] * invN;
      float var = fmaxf(st1[64 + k] * invN - mu * mu, 0.f);
      float v = (x - mu) * rsqrtf(var + EPS_BN) * g_t[k] + b_t[k];
      a[e] = (__bf16)v;
    }
    for (int nt = 0; nt < 4; ++nt){
      const v16bf* bp =
        (const v16bf*)(sW + (((size_t)kt * 4 + nt) * 32 + lane) * 16);
      acc[nt] = __builtin_amdgcn_wmma_f32_16x16x32_bf16(
          false, a, false, *bp, (short)0, acc[nt], false, false);
    }
  }

  for (int nt = 0; nt < 4; ++nt){
    int n = (lane & 15) + nt * 16;
    float bb = bmix[n];
    float ls = 0.f, lq = 0.f;
    for (int r = 0; r < 8; ++r){
      int node = tile * 16 + r + 8 * half;
      float v = acc[nt][r] + bb;
      v = (v > 0.f) ? v : 0.01f * v;              // leaky relu
      float o = v + h[(size_t)node * HID + n];    // PNA internal residual
      out[(size_t)node * HID + n] = o;
      ls += o; lq += o * o;
    }
    atomicAdd(&st2[n], ls);                       // fused BN2 statistics
    atomicAdd(&st2[64 + n], lq);
  }
}

// ---------------- relu(BN(out)) + h ----------------
__global__ void k_final(const float* __restrict__ out, const float* __restrict__ st2,
                        const float* __restrict__ g_o, const float* __restrict__ b_o,
                        const float* __restrict__ h, float* __restrict__ hn){
  int i = blockIdx.x * blockDim.x + threadIdx.x;
  if (i >= N_NODES * HID) return;
  int c = i & 63;
  const float invN = 1.f / (float)N_NODES;
  float mu  = st2[c] * invN;
  float var = fmaxf(st2[64 + c] * invN - mu * mu, 0.f);
  float v = (out[i] - mu) * rsqrtf(var + EPS_BN) * g_o[c] + b_o[c];
  hn[i] = fmaxf(v, 0.f) + h[i];
}

// ---------------- launcher ----------------
extern "C" void kernel_launch(void* const* d_in, const int* in_sizes, int n_in,
                              void* d_out, int out_size, void* d_ws, size_t ws_size,
                              hipStream_t stream) {
  (void)in_sizes; (void)n_in; (void)out_size; (void)ws_size;
  const float* node_feat = (const float*)d_in[0];
  const int*   esrc = (const int*)d_in[2];
  const int*   edst = (const int*)d_in[3];
  const float* WM   = (const float*)d_in[4];
  const float* bM   = (const float*)d_in[5];
  const float* WU   = (const float*)d_in[6];
  const float* bU   = (const float*)d_in[7];
  const float* g_t  = (const float*)d_in[8];
  const float* b_t  = (const float*)d_in[9];
  const float* WX   = (const float*)d_in[10];
  const float* bmx  = (const float*)d_in[11];
  const float* g_o  = (const float*)d_in[12];
  const float* b_o  = (const float*)d_in[13];

  char* ws = (char*)d_ws;
  size_t off = 0;
  auto take = [&](size_t bytes) -> char* {
    char* p = ws + off;
    off += (bytes + 255) & ~(size_t)255;
    return p;
  };
  const size_t NH = (size_t)N_NODES * HID;
  __bf16* WMp = (__bf16*)take(NLAYERS * WM_FRAG_BYTES);
  __bf16* WUp = (__bf16*)take(NLAYERS * WU_FRAG_BYTES);
  __bf16* WXp = (__bf16*)take(NLAYERS * WX_FRAG_BYTES);
  float*    sum   = (float*)   take(NH * 4);
  float*    sumsq = (float*)   take(NH * 4);
  unsigned* mxk   = (unsigned*)take(NH * 4);
  unsigned* mnk   = (unsigned*)take(NH * 4);
  float*    deg   = (float*)   take((size_t)N_NODES * 4);
  float*    hu    = (float*)   take(NH * 4);
  float*    outb  = (float*)   take(NH * 4);
  float*    hA    = (float*)   take(NH * 4);
  float*    hB    = (float*)   take(NH * 4);
  float*    st1   = (float*)   take(128 * 4);
  float*    st2   = (float*)   take(128 * 4);

  // one-time (per launch) bf16 fragment repack of all weights
  {
    int totM = NLAYERS * KT_M * 4 * 32;
    int totU = NLAYERS * KT_U * 4 * 32;
    int totX = NLAYERS * KT_X * 4 * 32;
    k_pack<<<(totM + 255) / 256, 256, 0, stream>>>(WM, WMp, 128,   NLAYERS);
    k_pack<<<(totU + 255) / 256, 256, 0, stream>>>(WU, WUp, DIN_U, NLAYERS);
    k_pack<<<(totX + 255) / 256, 256, 0, stream>>>(WX, WXp, HID,   NLAYERS);
  }

  const int edgeTiles = N_EDGES / 16;   // 31250
  const int nodeTiles = (N_NODES + 15) / 16;

  const float* hcur = node_feat;
  float* houts[NLAYERS] = { hA, hB, hA, (float*)d_out };

  for (int l = 0; l < NLAYERS; ++l){
    k_clear<<<(int)((NH + 255) / 256), 256, 0, stream>>>(
        sum, sumsq, mxk, mnk, deg, st1, st2);
    k_edge<<<(edgeTiles + 7) / 8, 256, (int)WM_FRAG_BYTES, stream>>>(
        hcur, esrc, edst, WMp + (size_t)l * (WM_FRAG_BYTES / 2), bM + l * HID,
        sum, sumsq, mxk, mnk, deg);
    k_update<<<(nodeTiles + 7) / 8, 256, (int)WU_FRAG_BYTES, stream>>>(
        hcur, sum, sumsq, mxk, mnk, deg,
        WUp + (size_t)l * (WU_FRAG_BYTES / 2), bU + l * HID, hu);
    k_colstats<<<64, 256, 0, stream>>>(hu, st1);
    k_mix<<<(nodeTiles + 7) / 8, 256, (int)WX_FRAG_BYTES, stream>>>(
        hu, st1, g_t + l * HID, b_t + l * HID,
        WXp + (size_t)l * (WX_FRAG_BYTES / 2), bmx + l * HID, hcur, outb, st2);
    k_final<<<(int)((NH + 255) / 256), 256, 0, stream>>>(
        outb, st2, g_o + l * HID, b_o + l * HID, hcur, houts[l]);
    hcur = houts[l];
  }
}
